// SE3Net_11115375362057
// MI455X (gfx1250) — matrix operand order repaired and checked by reading, not compiled, not run on hardware
//
#include <hip/hip_runtime.h>
#include <cstring>

typedef _Float16 v16h __attribute__((ext_vector_type(16)));
typedef _Float16 v8h  __attribute__((ext_vector_type(8)));
typedef float    v8f  __attribute__((ext_vector_type(8)));

#define NPAIR 32768   // B*N*N = 8*64*64
#define NSITE 512     // B*N
#define LDH   128     // padded hidden dim (H=100 -> 128)

// ---------------------------------------------------------------------------
// Metadata structs (shared host/device, passed by value as kernel args)
// ---------------------------------------------------------------------------
struct ConvMeta {
  int nIn, nOut, dimIn, dimU;
  int inMul[4], inL[4], inOff[4];
  int outMul[6], outL[6], outOff[6];
  int iwStart[6][4];
  int off3j[4][4][4];
};
struct GateMeta {
  int nOut, dimU, dimOut;
  int mul[4], l[4], off[4], goff[4];
};

// ---------------------------------------------------------------------------
// C3J (real Gaunt-like coefficient) computation on device
// ---------------------------------------------------------------------------
__device__ double d_fact(int n){ double r=1.0; for(int i=2;i<=n;++i) r*=(double)i; return r; }

__device__ double d_w3j(int j1,int j2,int j3,int m1,int m2,int m3){
  if (m1+m2+m3 != 0) return 0.0;
  int dj = j1-j2; if (dj<0) dj=-dj;
  if (j3 < dj || j3 > j1+j2) return 0.0;
  if (m1> j1||m1<-j1||m2>j2||m2<-j2||m3>j3||m3<-j3) return 0.0;
  double t = sqrt(d_fact(j1+j2-j3)*d_fact(j1-j2+j3)*d_fact(-j1+j2+j3)/d_fact(j1+j2+j3+1));
  t *= sqrt(d_fact(j1+m1)*d_fact(j1-m1)*d_fact(j2+m2)*d_fact(j2-m2)*d_fact(j3+m3)*d_fact(j3-m3));
  double s = 0.0;
  for (int k=0;k<=j1+j2-j3;++k){
    int d0=k, d1=j1+j2-j3-k, d2=j1-m1-k, d3=j2+m2-k, d4=j3-j2+m1+k, d5=j3-j1-m2+k;
    if (d0<0||d1<0||d2<0||d3<0||d4<0||d5<0) continue;
    double den = d_fact(d0)*d_fact(d1)*d_fact(d2)*d_fact(d3)*d_fact(d4)*d_fact(d5);
    s += ((k&1)? -1.0:1.0)/den;
  }
  double sign = (((j1-j2-m3)&1)!=0)? -1.0:1.0;
  return sign*t*s;
}

// Row a of the complex unitary U(l): at most 2 nonzero columns.
__device__ int d_Urow(int l,int a,int* cols,double* re,double* im){
  const double s = 0.70710678118654752440;
  int m = a - l;
  if (m > 0){
    cols[0]=l+m; re[0]=((m&1)? -1.0:1.0)*s; im[0]=0.0;
    cols[1]=l-m; re[1]=s;                   im[1]=0.0;
    return 2;
  } else if (m < 0){
    int am = -m;
    cols[0]=l+m; re[0]=0.0; im[0]=s;
    cols[1]=l-m; re[1]=0.0; im[1]= -(((am&1))? -1.0:1.0)*s;
    return 2;
  }
  cols[0]=l; re[0]=1.0; im[0]=0.0;
  return 1;
}

__global__ void c3j_kernel(float* __restrict__ out){
  __shared__ float sre[343], sim[343];
  __shared__ int   s_useRe;
  __shared__ float s_norm;
  int tid = threadIdx.x;
  int off = 0;
  for (int lo=0; lo<4; ++lo)
  for (int li=0; li<4; ++li){
    int lfmin = (lo>li)? lo-li : li-lo;
    int lfmax = (lo+li<3)? lo+li : 3;
    for (int lf=lfmin; lf<=lfmax; ++lf){
      int no=2*lo+1, ni=2*li+1, nf=2*lf+1, ne=no*ni*nf;
      for (int e=tid; e<ne; e+=blockDim.x){
        int a = e/(ni*nf), b=(e/nf)%ni, c=e%nf;
        int c1[2],c2[2],c3c[2];
        double r1[2],i1[2],r2[2],i2[2],r3[2],i3[2];
        int n1=d_Urow(lo,a,c1,r1,i1), n2=d_Urow(li,b,c2,r2,i2), n3=d_Urow(lf,c,c3c,r3,i3);
        double qr=0.0, qi=0.0;
        for (int t1=0;t1<n1;++t1)
        for (int t2=0;t2<n2;++t2)
        for (int t3=0;t3<n3;++t3){
          double wj = d_w3j(lo,li,lf, c1[t1]-lo, c2[t2]-li, c3c[t3]-lf);
          if (wj==0.0) continue;
          double ar = r1[t1]*r2[t2]-i1[t1]*i2[t2];
          double ai = r1[t1]*i2[t2]+i1[t1]*r2[t2];
          double br = ar*r3[t3]-ai*i3[t3];
          double bi = ar*i3[t3]+ai*r3[t3];
          qr += br*wj; qi += bi*wj;
        }
        sre[e]=(float)qr; sim[e]=(float)qi;
      }
      __syncthreads();
      if (tid==0){
        float mr=0.f, mi=0.f;
        for (int e=0;e<ne;++e){ mr=fmaxf(mr,fabsf(sre[e])); mi=fmaxf(mi,fabsf(sim[e])); }
        int ur = (mr>=mi)?1:0;
        float nn=0.f;
        for (int e=0;e<ne;++e){ float v = ur? sre[e]:sim[e]; nn += v*v; }
        s_useRe = ur;
        s_norm  = (nn>0.f)? sqrtf(nn) : 1.f;
      }
      __syncthreads();
      for (int e=tid; e<ne; e+=blockDim.x)
        out[off+e] = (s_useRe? sre[e] : sim[e]) / s_norm;
      __syncthreads();
      off += ne;
    }
  }
}

// ---------------------------------------------------------------------------
// Weight transposes (row-major (k,o) -> padded out-major (o,k) f16)
// ---------------------------------------------------------------------------
__global__ void transW_kernel(const float* __restrict__ w, _Float16* __restrict__ wT, int nLayers){
  int idx = blockIdx.x*blockDim.x + threadIdx.x;
  int tot = nLayers*16384;
  if (idx >= tot) return;
  int layer = idx >> 14, rem = idx & 16383, o = rem >> 7, k = rem & 127;
  float v = (o<100 && k<100) ? w[layer*10000 + k*100 + o] : 0.f;
  wT[idx] = (_Float16)v;
}
__global__ void transWout_kernel(const float* __restrict__ w, _Float16* __restrict__ wT, int nw, int nwPad){
  int idx = blockIdx.x*blockDim.x + threadIdx.x;
  int tot = nwPad*128;
  if (idx >= tot) return;
  int n = idx >> 7, k = idx & 127;
  float v = (n<nw && k<100) ? w[k*nw + n] : 0.f;
  wT[idx] = (_Float16)v;
}

// ---------------------------------------------------------------------------
// Radial basis + first MLP layer: h0 = relu(basis @ w_in / sqrt(3))
// ---------------------------------------------------------------------------
__global__ __launch_bounds__(256)
void radial_kernel(const float* __restrict__ geom, const float* __restrict__ w_in,
                   _Float16* __restrict__ h0){
  int p = blockIdx.x*256 + threadIdx.x;      // pair id: b*4096 + n*64 + j
  int b = p >> 12, n = (p >> 6) & 63, j = p & 63;
  float dx = geom[(b*64+n)*3+0] - geom[(b*64+j)*3+0];
  float dy = geom[(b*64+n)*3+1] - geom[(b*64+j)*3+1];
  float dz = geom[(b*64+n)*3+2] - geom[(b*64+j)*3+2];
  float dist = sqrtf(dx*dx+dy*dy+dz*dz + 1e-12f);
  float bas[3];
  #pragma unroll
  for (int c=0;c<3;++c){
    float d = (dist - 1.5f*(float)c) * (1.f/1.5f);
    float cv = cosf(1.5707963267948966f*d);
    bas[c] = (fabsf(d) < 1.f) ? cv*cv : 0.f;
  }
  const float s = 0.5773502691896258f;  // 1/sqrt(3)
  _Float16* hr = h0 + (size_t)p*LDH;
  for (int k=0;k<100;++k){
    float v = (bas[0]*w_in[k] + bas[1]*w_in[100+k] + bas[2]*w_in[200+k]) * s;
    hr[k] = (_Float16)(v>0.f? v : 0.f);
  }
  for (int k=100;k<LDH;++k) hr[k] = (_Float16)0.f;
}

// ---------------------------------------------------------------------------
// Persistent WMMA MLP: 64-row tile per workgroup, 49 layers LDS-resident
// ---------------------------------------------------------------------------
__global__ __launch_bounds__(256)
void mlp_kernel(const _Float16* __restrict__ h0, const _Float16* __restrict__ wT,
                _Float16* __restrict__ hF, int nLayers){
  __shared__ __align__(64) _Float16 sh[2*64*LDH + 128*LDH];  // 16K + 16K + 32K bytes
  _Float16* hA = sh;
  _Float16* hB = sh + 64*LDH;
  _Float16* wb = sh + 2*64*LDH;
  int tid = threadIdx.x;
  size_t rowbase = (size_t)blockIdx.x * 64 * LDH;
  { // stage my 64 activation rows into LDS
    const float4* g4 = (const float4*)(h0 + rowbase);
    float4* s4 = (float4*)hA;
    for (int i=tid; i<1024; i+=256) s4[i] = g4[i];
  }
  int wave = tid >> 5, lane = tid & 31, l16 = lane & 15, hi = lane >> 4;
  _Float16* hin = hA; _Float16* hout = hB;
  for (int layer=0; layer<nLayers; ++layer){
    __syncthreads();                         // prior reads of wb / hin done
    { // cooperative load of this layer's transposed weights
      const float4* wl = (const float4*)(wT + (size_t)layer*16384);
      float4* wd = (float4*)wb;
      for (int i=tid; i<2048; i+=256) wd[i] = wl[i];
    }
    if (layer+1 < nLayers)                   // pull next layer toward L2/L0
      __builtin_prefetch(wT + (size_t)(layer+1)*16384 + tid*64, 0, 0);
    __syncthreads();
    // 4 (M) x 7 (N) output tiles of 16x16 over 8 waves
    for (int t=wave; t<28; t+=8){
      int msub = t & 3, nsub = t >> 2;
      v8f acc = {0.f,0.f,0.f,0.f,0.f,0.f,0.f,0.f};
      const _Float16* ar = hin + (msub*16 + l16)*LDH;
      const _Float16* br = wb  + (nsub*16 + l16)*LDH;
      #pragma unroll
      for (int kk=0; kk<LDH; kk+=32){
        v8h alo = *(const v8h*)(ar + kk + hi*8);
        v8h ahi = *(const v8h*)(ar + kk + hi*8 + 16);
        v16h a = __builtin_shufflevector(alo, ahi, 0,1,2,3,4,5,6,7,8,9,10,11,12,13,14,15);
        v8h blo = *(const v8h*)(br + kk + hi*8);
        v8h bhi = *(const v8h*)(br + kk + hi*8 + 16);
        v16h bfrag = __builtin_shufflevector(blo, bhi, 0,1,2,3,4,5,6,7,8,9,10,11,12,13,14,15);
        acc = __builtin_amdgcn_wmma_f32_16x16x32_f16(false, a, false, bfrag,
                                                     (short)0, acc, false, false);
      }
      _Float16* cb = hout + (msub*16 + hi*8)*LDH + nsub*16 + l16;
      #pragma unroll
      for (int r=0;r<8;++r){
        float v = acc[r]*0.1f;               // / sqrt(H=100)
        cb[r*LDH] = (_Float16)(v>0.f? v : 0.f);
      }
    }
    __syncthreads();
    _Float16* tmp = hin; hin = hout; hout = tmp;
  }
  { // write final activations back
    float4* o4 = (float4*)(hF + rowbase);
    const float4* f4 = (const float4*)hin;
    for (int i=tid; i<1024; i+=256) o4[i] = f4[i];
  }
}

// ---------------------------------------------------------------------------
// R = hF @ w_out / sqrt(H)   (WMMA GEMM, 64x64 tile per workgroup)
// ---------------------------------------------------------------------------
__global__ __launch_bounds__(256)
void rgemm_kernel(const _Float16* __restrict__ hF, const _Float16* __restrict__ wOutT,
                  _Float16* __restrict__ R, int nwPad){
  __shared__ __align__(64) _Float16 sA[64*LDH];
  __shared__ __align__(64) _Float16 sB[64*LDH];
  int tid = threadIdx.x;
  size_t rowbase = (size_t)blockIdx.x * 64;
  {
    const float4* ga = (const float4*)(hF + rowbase*LDH);
    const float4* gb = (const float4*)(wOutT + (size_t)blockIdx.y*64*LDH);
    float4* a4=(float4*)sA; float4* b4=(float4*)sB;
    for (int i=tid; i<1024; i+=256){ a4[i]=ga[i]; b4[i]=gb[i]; }
  }
  __syncthreads();
  int wave = tid >> 5, lane = tid & 31, l16 = lane & 15, hi = lane >> 4;
  for (int t=wave; t<16; t+=8){
    int msub = t & 3, nsub = t >> 2;
    v8f acc = {0.f,0.f,0.f,0.f,0.f,0.f,0.f,0.f};
    const _Float16* ar = sA + (msub*16 + l16)*LDH;
    const _Float16* br = sB + (nsub*16 + l16)*LDH;
    #pragma unroll
    for (int kk=0; kk<LDH; kk+=32){
      v8h alo = *(const v8h*)(ar + kk + hi*8);
      v8h ahi = *(const v8h*)(ar + kk + hi*8 + 16);
      v16h a = __builtin_shufflevector(alo, ahi, 0,1,2,3,4,5,6,7,8,9,10,11,12,13,14,15);
      v8h blo = *(const v8h*)(br + kk + hi*8);
      v8h bhi = *(const v8h*)(br + kk + hi*8 + 16);
      v16h bfrag = __builtin_shufflevector(blo, bhi, 0,1,2,3,4,5,6,7,8,9,10,11,12,13,14,15);
      acc = __builtin_amdgcn_wmma_f32_16x16x32_f16(false, a, false, bfrag,
                                                   (short)0, acc, false, false);
    }
    int col = blockIdx.y*64 + nsub*16 + l16;
    #pragma unroll
    for (int r=0;r<8;++r){
      size_t row = rowbase + msub*16 + hi*8 + r;
      R[row*(size_t)nwPad + col] = (_Float16)(acc[r]*0.1f);   // no relu on w_out
    }
  }
}

// ---------------------------------------------------------------------------
// Equivariant tensor-product convolution (per output site)
// ---------------------------------------------------------------------------
__global__ __launch_bounds__(128)
void conv_kernel(const float* __restrict__ fin, const _Float16* __restrict__ R,
                 const float* __restrict__ c3j, const float* __restrict__ geom,
                 float* __restrict__ uout, int nwPad, ConvMeta M){
  __shared__ float sF[64*64];
  __shared__ float sY[64*16];
  int site = blockIdx.x;            // b*64 + n
  int b = site >> 6, n = site & 63;
  int tid = threadIdx.x;
  int tot = 64*M.dimIn;
  for (int i=tid; i<tot; i+=blockDim.x){
    int j = i / M.dimIn, k = i % M.dimIn;
    sF[j*M.dimIn + k] = 0.5f * fin[((size_t)(b*64+j))*M.dimIn + k];  // f / sqrt(4)
  }
  if (tid < 64){
    int j = tid;
    float x = geom[(b*64+n)*3+0] - geom[(b*64+j)*3+0];
    float y = geom[(b*64+n)*3+1] - geom[(b*64+j)*3+1];
    float z = geom[(b*64+n)*3+2] - geom[(b*64+j)*3+2];
    float dist = sqrtf(x*x+y*y+z*z + 1e-12f);
    float inv = 1.f/dist; x*=inv; y*=inv; z*=inv;
    float r2 = x*x+y*y+z*z;
    float* Y = sY + j*16;
    Y[0]=1.f;
    Y[1]=1.7320508f*y;  Y[2]=1.7320508f*z;  Y[3]=1.7320508f*x;
    Y[4]=3.8729833f*x*y; Y[5]=3.8729833f*y*z; Y[6]=1.1180340f*(3.f*z*z-r2);
    Y[7]=3.8729833f*x*z; Y[8]=1.9364917f*(x*x-y*y);
    Y[9]=2.0916501f*y*(3.f*x*x-y*y); Y[10]=10.2469508f*x*y*z;
    Y[11]=1.6201852f*y*(5.f*z*z-r2); Y[12]=1.3228757f*z*(5.f*z*z-3.f*r2);
    Y[13]=1.6201852f*x*(5.f*z*z-r2); Y[14]=5.1234754f*z*(x*x-y*y);
    Y[15]=2.0916501f*x*(x*x-3.f*y*y);
  }
  __syncthreads();
  for (int e=tid; e<M.dimU; e+=blockDim.x){
    int oe=0, mo=1, lo=0, u=0, o=0;
    for (int i=0;i<M.nOut;++i){
      int d = M.outMul[i]*(2*M.outL[i]+1);
      if (e >= M.outOff[i] && e < M.outOff[i]+d){
        oe=i; mo=M.outMul[i]; lo=M.outL[i];
        u=(e-M.outOff[i])/(2*lo+1); o=(e-M.outOff[i])%(2*lo+1);
        break;
      }
    }
    float acc = 0.f;
    for (int j=0;j<64;++j){
      const _Float16* Rr = R + ((size_t)site*64 + j)*(size_t)nwPad;
      const float* Yj = sY + j*16;
      const float* Fj = sF + j*M.dimIn;
      for (int ie=0; ie<M.nIn; ++ie){
        int mi = M.inMul[ie], li = M.inL[ie];
        int lf0 = (lo>li)? lo-li : li-lo;
        int lf1 = (lo+li<3)? lo+li : 3;
        int nlf = lf1-lf0+1;
        int ni = 2*li+1;
        float pacc = 0.f;
        for (int t=0;t<nlf;++t){
          int lf = lf0+t, nf = 2*lf+1;
          const float* C  = c3j + M.off3j[lo][li][lf];
          const float* Yl = Yj + lf*lf;
          float Ai[7];
          for (int i2=0;i2<ni;++i2){
            float s=0.f;
            const float* Cr = C + (o*ni + i2)*nf;
            for (int m=0;m<nf;++m) s += Yl[m]*Cr[m];
            Ai[i2]=s;
          }
          int iw = M.iwStart[oe][ie] + t*mo*mi + u*mi;
          const float* Fv = Fj + M.inOff[ie];
          float s2=0.f;
          for (int v=0;v<mi;++v){
            float fv=0.f;
            for (int i2=0;i2<ni;++i2) fv += Ai[i2]*Fv[v*ni+i2];
            s2 += (float)Rr[iw+v]*fv;
          }
          pacc += s2;
        }
        acc += pacc * rsqrtf((float)(mi*nlf));
      }
    }
    uout[(size_t)site*M.dimU + e] = acc * 0.125f;    // / sqrt(N=64)
  }
}

// ---------------------------------------------------------------------------
// Gating
// ---------------------------------------------------------------------------
__global__ void gate_kernel(const float* __restrict__ u, float* __restrict__ fo, GateMeta g){
  int idx = blockIdx.x*blockDim.x + threadIdx.x;
  if (idx >= NSITE*g.dimOut) return;
  int site = idx / g.dimOut, e = idx % g.dimOut;
  const float* ur = u + (size_t)site*g.dimU;
  float val = 0.f;
  for (int i=0;i<g.nOut;++i){
    int m=g.mul[i], l=g.l[i], off=g.off[i], d=m*(2*l+1);
    if (e >= off && e < off+d){
      float x = ur[e];
      if (l==0){ val = x>0.f? x : 0.f; }
      else {
        int uu = (e-off)/(2*l+1);
        float gv = ur[g.dimOut + g.goff[i] + uu];
        val = x / (1.f + __expf(-gv));
      }
      break;
    }
  }
  fo[(size_t)site*g.dimOut + e] = val;
}

// ---------------------------------------------------------------------------
// Head: mean over nodes + linear classifier
// ---------------------------------------------------------------------------
__global__ void head_kernel(const float* __restrict__ f, const float* __restrict__ lw,
                            const float* __restrict__ lb, float* __restrict__ out){
  int idx = threadIdx.x;
  if (idx >= 80) return;
  int b = idx/10, c = idx%10;
  float s = 0.f;
  for (int n=0;n<64;++n){
    const float* fr = f + ((size_t)(b*64+n))*64;
    float t = 0.f;
    for (int k=0;k<64;++k) t += fr[k]*lw[k*10+c];
    s += t;
  }
  out[idx] = s*(1.f/64.f) + lb[c];
}

// ---------------------------------------------------------------------------
// Host-side metadata builders
// ---------------------------------------------------------------------------
static const int RSmul[5][4] = {{1,0,0,0},{2,2,2,1},{4,4,4,4},{6,4,4,0},{64,0,0,0}};
static const int RSl  [5][4] = {{0,0,0,0},{0,1,2,3},{0,1,2,3},{0,1,2,0},{0,0,0,0}};
static const int RSn  [5]    = {1,4,4,3,1};

static inline int lfcnt(int lo,int li){
  int a = (lo>li)? lo-li : li-lo;
  int b = (lo+li<3)? lo+li : 3;
  return b-a+1;
}

static ConvMeta buildConv(int bi){
  ConvMeta m; memset(&m,0,sizeof(m));
  m.nIn = RSn[bi];
  int off=0;
  for (int i=0;i<m.nIn;++i){
    m.inMul[i]=RSmul[bi][i]; m.inL[i]=RSl[bi][i]; m.inOff[i]=off;
    off += m.inMul[i]*(2*m.inL[i]+1);
  }
  m.dimIn = off;
  int no=0, uoff=0, gates=0;
  for (int i=0;i<RSn[bi+1];++i){
    int mo=RSmul[bi+1][i], lo=RSl[bi+1][i];
    m.outMul[no]=mo; m.outL[no]=lo; m.outOff[no]=uoff;
    uoff += mo*(2*lo+1);
    if (lo>0) gates += mo;
    ++no;
  }
  if (gates>0){ m.outMul[no]=gates; m.outL[no]=0; m.outOff[no]=uoff; uoff+=gates; ++no; }
  m.nOut=no; m.dimU=uoff;
  int iw=0;
  for (int oe=0;oe<no;++oe)
    for (int ie=0;ie<m.nIn;++ie){
      m.iwStart[oe][ie]=iw;
      iw += m.outMul[oe]*m.inMul[ie]*lfcnt(m.outL[oe], m.inL[ie]);
    }
  int o3=0;
  for (int lo=0;lo<4;++lo) for (int li=0;li<4;++li){
    int a=(lo>li)?lo-li:li-lo, b=(lo+li<3)?lo+li:3;
    for (int lf=a;lf<=b;++lf){ m.off3j[lo][li][lf]=o3; o3 += (2*lo+1)*(2*li+1)*(2*lf+1); }
  }
  return m;
}

static GateMeta buildGate(int bi, int dimU){
  GateMeta g; memset(&g,0,sizeof(g));
  g.nOut = RSn[bi+1]; g.dimU = dimU;
  int off=0, gi=0;
  for (int i=0;i<g.nOut;++i){
    g.mul[i]=RSmul[bi+1][i]; g.l[i]=RSl[bi+1][i]; g.off[i]=off; g.goff[i]=gi;
    off += g.mul[i]*(2*g.l[i]+1);
    if (g.l[i]>0) gi += g.mul[i];
  }
  g.dimOut = off;
  return g;
}

// ---------------------------------------------------------------------------
// Launch
// ---------------------------------------------------------------------------
extern "C" void kernel_launch(void* const* d_in, const int* in_sizes, int n_in,
                              void* d_out, int out_size, void* d_ws, size_t ws_size,
                              hipStream_t stream){
  (void)n_in; (void)out_size; (void)ws_size;

  // Input ordering: dict insertion order (w_in first) vs pytree-sorted (lin_b at [2]).
  const float* features; const float* geometry; const float* lin_w; const float* lin_b;
  const float* w_in_p[4]; const float* w_hid_p[4]; const float* w_out_p[4];
  int nL_p[4], nw_p[4];
  features = (const float*)d_in[0];
  geometry = (const float*)d_in[1];
  if (in_sizes[2] == 300){           // insertion order: params then lin_w, lin_b
    for (int bi=0;bi<4;++bi){
      w_in_p[bi]  = (const float*)d_in[2+3*bi];
      w_hid_p[bi] = (const float*)d_in[3+3*bi];
      w_out_p[bi] = (const float*)d_in[4+3*bi];
      nL_p[bi] = in_sizes[3+3*bi]/10000;
      nw_p[bi] = in_sizes[4+3*bi]/100;
    }
    lin_w = (const float*)d_in[14];
    lin_b = (const float*)d_in[15];
  } else {                           // sorted keys: lin_b, lin_w, then (w_hid,w_in,w_out)x4
    lin_b = (const float*)d_in[2];
    lin_w = (const float*)d_in[3];
    for (int bi=0;bi<4;++bi){
      w_hid_p[bi] = (const float*)d_in[4+3*bi];
      w_in_p[bi]  = (const float*)d_in[5+3*bi];
      w_out_p[bi] = (const float*)d_in[6+3*bi];
      nL_p[bi] = in_sizes[4+3*bi]/10000;
      nw_p[bi] = in_sizes[6+3*bi]/100;
    }
  }

  char* base = (char*)d_ws;
  size_t cur = 0;
  auto alloc = [&](size_t bytes)->void*{
    void* p = base + cur;
    cur += (bytes + 255) & ~(size_t)255;
    return p;
  };
  float*    c3j   = (float*)   alloc(3436*sizeof(float));
  _Float16* h0    = (_Float16*)alloc((size_t)NPAIR*LDH*2);
  _Float16* hF    = (_Float16*)alloc((size_t)NPAIR*LDH*2);
  _Float16* wT    = (_Float16*)alloc((size_t)64*16384*2);
  _Float16* wOutT = (_Float16*)alloc((size_t)896*128*2);
  _Float16* Rbuf  = (_Float16*)alloc((size_t)NPAIR*896*2);
  float*    ubuf  = (float*)   alloc((size_t)NSITE*96*4);
  float*    fA    = (float*)   alloc((size_t)NSITE*64*4);
  float*    fB    = (float*)   alloc((size_t)NSITE*64*4);

  c3j_kernel<<<1,256,0,stream>>>(c3j);

  const float* fin = features;
  for (int bi=0; bi<4; ++bi){
    int nLayers = nL_p[bi];
    int nw      = nw_p[bi];
    int nwPad   = (nw + 63) & ~63;

    transW_kernel   <<<(nLayers*16384+255)/256,256,0,stream>>>(w_hid_p[bi], wT, nLayers);
    transWout_kernel<<<(nwPad*128+255)/256,   256,0,stream>>>(w_out_p[bi], wOutT, nw, nwPad);
    radial_kernel   <<<NPAIR/256,             256,0,stream>>>(geometry, w_in_p[bi], h0);
    mlp_kernel      <<<NPAIR/64,              256,0,stream>>>(h0, wT, hF, nLayers);
    rgemm_kernel    <<<dim3(NPAIR/64, nwPad/64),256,0,stream>>>(hF, wOutT, Rbuf, nwPad);

    ConvMeta cm = buildConv(bi);
    conv_kernel<<<NSITE,128,0,stream>>>(fin, Rbuf, c3j, geometry, ubuf, nwPad, cm);
    GateMeta gm = buildGate(bi, cm.dimU);
    float* fout = (bi & 1) ? fB : fA;
    gate_kernel<<<(NSITE*gm.dimOut+255)/256,256,0,stream>>>(ubuf, fout, gm);
    fin = fout;
  }
  head_kernel<<<1,128,0,stream>>>(fin, lin_w, lin_b, (float*)d_out);
}
